// RandomCoordination_84593675862310
// MI455X (gfx1250) — matrix operand order repaired and checked by reading, not compiled, not run on hardware
//
#include <hip/hip_runtime.h>

typedef __attribute__((ext_vector_type(16))) _Float16 v16h;
typedef __attribute__((ext_vector_type(8)))  float    v8f;

#define BB 8
#define NN 2048
#define DD 128
#define KK 16

// Monotone 64-bit key: (order-preserving float bits << 32) | ~index.
// Max key == max value, ties broken toward the smaller index (matches lax.top_k).
__device__ __forceinline__ unsigned long long rc_key(float f, int j) {
  unsigned u = __float_as_uint(f);
  unsigned m = u ^ (unsigned)(((int)u >> 31) | 0x80000000);
  return (((unsigned long long)m) << 32) | (unsigned)(~j);
}

__global__ __launch_bounds__(512)
void rc_kernel(const float* __restrict__ s,
               const float* __restrict__ noise,
               const int* __restrict__ adj,      // adj_allowed
               const int* __restrict__ activem,  // active_mask
               const int* __restrict__ actm,     // act_mask
               float* __restrict__ ctx,
               float* __restrict__ gate,
               float* __restrict__ wout)
{
  extern __shared__ char smem_raw[];
  float*    sc     = (float*)smem_raw;            // [16][NN] masked scores
  unsigned* selidx = (unsigned*)(sc + 16 * NN);   // [16][16] selected j (0 if invalid)
  unsigned* vmask  = selidx + 16 * 16;            // [16] valid-slot bitmask
  float*    wvals  = (float*)(vmask + 16);        // [16] 1/max(cnt,1)

  const int tid   = (int)threadIdx.x;
  const int lane  = tid & 31;
  const int wv    = tid >> 5;           // wave 0..15
  const int tile  = (int)blockIdx.x;    // 0..127
  const int b     = (int)blockIdx.y;    // 0..7
  const int ibase = tile * 16;

  // ---------------- Phase 1: per-wave top-16 for row r = wv ----------------
  {
    const int r = wv;
    const int i = ibase + r;
    const size_t rowoff = (size_t)(b * NN + i) * NN;
    const float* nrow = noise + rowoff;
    const int*   arow = adj + rowoff;
    const int*   am   = activem + (size_t)b * NN;
    float* scr = sc + r * NN;

    #pragma unroll
    for (int t = 0; t < 16; ++t) {
      const int e = (lane + 32 * t) * 4;
      const float4 nv = *(const float4*)(nrow + e);
      const int4   av = *(const int4*)(arow + e);
      const int4   mv = *(const int4*)(am + e);
      float4 scv;
      scv.x = (mv.x > 0 && av.x > 0) ? nv.x : -1.0f;
      scv.y = (mv.y > 0 && av.y > 0) ? nv.y : -1.0f;
      scv.z = (mv.z > 0 && av.z > 0) ? nv.z : -1.0f;
      scv.w = (mv.w > 0 && av.w > 0) ? nv.w : -1.0f;
      *(float4*)(scr + e) = scv;
    }

    unsigned my_selj  = 0u;
    int      my_selok = 0;

    for (int rd = 0; rd < KK; ++rd) {
      unsigned long long best = 0ull;
      #pragma unroll
      for (int t = 0; t < 16; ++t) {
        const int e = (lane + 32 * t) * 4;
        const float4 v = *(const float4*)(scr + e);
        unsigned long long k0 = rc_key(v.x, e + 0);
        unsigned long long k1 = rc_key(v.y, e + 1);
        unsigned long long k2 = rc_key(v.z, e + 2);
        unsigned long long k3 = rc_key(v.w, e + 3);
        k0 = k0 > k1 ? k0 : k1;
        k2 = k2 > k3 ? k2 : k3;
        k0 = k0 > k2 ? k0 : k2;
        best = best > k0 ? best : k0;
      }
      #pragma unroll
      for (int off = 16; off > 0; off >>= 1) {
        unsigned long long o = __shfl_xor(best, off, 32);
        best = o > best ? o : best;
      }
      const unsigned j  = (~(unsigned)best) & (NN - 1);
      const int      ok = (int)(best >> 63);   // 1 iff value >= 0
      if (lane == rd) { my_selj = j; my_selok = ok; }
      if (lane == 0)  scr[j] = -2.0f;          // pop; DS ops in-order per wave
    }

    const int actr = actm[(size_t)b * NN + i] > 0;
    unsigned bits = (unsigned)__ballot(my_selok) & 0xFFFFu;
    if (!actr) bits = 0u;
    const int cnt = __popc(bits);
    const float wvf = 1.0f / (float)(cnt > 1 ? cnt : 1);
    // Pre-masked indices: invalid slots -> 0 (safe dummy gather row).
    if (lane < 16) selidx[r * 16 + lane] = ((bits >> lane) & 1u) ? my_selj : 0u;
    if (lane == 0) { vmask[r] = bits; wvals[r] = wvf; }
  }

  __syncthreads();

  if (wv < 8) {
    // ------------- Phase 2a: ctx tile via WMMA (waves 0..7) ---------------
    // ctx[16x128] = A[16x256] (block-diagonal weights) x Sg[256x128] (gathered
    // s rows); K chunked by 32 into v_wmma_f32_16x16x32_f16. Wave wv owns
    // output columns [16*wv, 16*wv+16).
    const int half  = lane >> 4;     // lane group
    const int ncol  = lane & 15;     // N column / A row M
    const int dcol  = wv * 16 + ncol;
    const float* sb = s + (size_t)b * NN * DD;
    v8f c = {0.f, 0.f, 0.f, 0.f, 0.f, 0.f, 0.f, 0.f};

    #pragma unroll
    for (int c8 = 0; c8 < 8; ++c8) {
      // B fragment: lanes 0-15 hold K=0..15 (src row 2*c8), lanes 16-31 hold
      // K=16..31 (src row 2*c8+1); VGPR n packs K=base+2n, base+2n+1.
      const int rp = 2 * c8 + half;
      const unsigned* sidx = selidx + rp * 16;
      const uint4 i0 = *(const uint4*)(sidx + 0);
      const uint4 i1 = *(const uint4*)(sidx + 4);
      const uint4 i2 = *(const uint4*)(sidx + 8);
      const uint4 i3 = *(const uint4*)(sidx + 12);
      v16h bf;
      bf[0]  = (_Float16)sb[(size_t)i0.x * DD + dcol];
      bf[1]  = (_Float16)sb[(size_t)i0.y * DD + dcol];
      bf[2]  = (_Float16)sb[(size_t)i0.z * DD + dcol];
      bf[3]  = (_Float16)sb[(size_t)i0.w * DD + dcol];
      bf[4]  = (_Float16)sb[(size_t)i1.x * DD + dcol];
      bf[5]  = (_Float16)sb[(size_t)i1.y * DD + dcol];
      bf[6]  = (_Float16)sb[(size_t)i1.z * DD + dcol];
      bf[7]  = (_Float16)sb[(size_t)i1.w * DD + dcol];
      bf[8]  = (_Float16)sb[(size_t)i2.x * DD + dcol];
      bf[9]  = (_Float16)sb[(size_t)i2.y * DD + dcol];
      bf[10] = (_Float16)sb[(size_t)i2.z * DD + dcol];
      bf[11] = (_Float16)sb[(size_t)i2.w * DD + dcol];
      bf[12] = (_Float16)sb[(size_t)i3.x * DD + dcol];
      bf[13] = (_Float16)sb[(size_t)i3.y * DD + dcol];
      bf[14] = (_Float16)sb[(size_t)i3.z * DD + dcol];
      bf[15] = (_Float16)sb[(size_t)i3.w * DD + dcol];

      // A fragment, fully branch-free. Per lane at most one of the two chunk
      // rows matches M, so collapse validity into one 16-bit mask following
      // the contiguous k order used here for A and B:
  //   K = t + 16*half (row 2*c8+half, bit t of vm0/vm1), matching B
      const int M = ncol;
      const unsigned vm0 = vmask[2 * c8];
      const unsigned vm1 = vmask[2 * c8 + 1];
      const float    wf0 = wvals[2 * c8];
      const float    wf1 = wvals[2 * c8 + 1];
      const unsigned m0 = (M == 2 * c8)     ? vm0 : 0u;   // flat cndmask
      const unsigned m1 = (M == 2 * c8 + 1) ? vm1 : 0u;   // flat cndmask
      const unsigned sh = 8u * (unsigned)half;
      const unsigned sel16 = (half ? m1 : m0) & 0xFFFFu;
      const _Float16 wh = (_Float16)((M == 2 * c8) ? wf0 : wf1);
      v16h af;
      #pragma unroll
      for (int t = 0; t < 16; ++t) {
        af[t] = ((sel16 >> t) & 1u) ? wh : (_Float16)0.f;  // single-cond select
      }
      c = __builtin_amdgcn_wmma_f32_16x16x32_f16(false, af, false, bf,
                                                 (short)0, c, false, false);
    }

    // D layout: VGPR v -> M = v + 8*half, N = lane%16
    float* cb = ctx + (size_t)(b * NN + ibase) * DD;
    #pragma unroll
    for (int v = 0; v < 8; ++v) {
      const int M = v + 8 * half;
      cb[(size_t)M * DD + wv * 16 + ncol] = c[v];
    }
  } else {
    // ------------- Phase 2b: gate / w rows (waves 8..15) ------------------
    // Wave q handles rows 2*(q-8) and 2*(q-8)+1: zero-fill then scatter.
    const int q = wv - 8;
    #pragma unroll
    for (int rr = 0; rr < 2; ++rr) {
      const int r = 2 * q + rr;
      const int i = ibase + r;
      const size_t rowoff = (size_t)(b * NN + i) * NN;
      float* grow = gate + rowoff;
      float* wrow = wout + rowoff;
      const float4 z4 = make_float4(0.f, 0.f, 0.f, 0.f);
      #pragma unroll
      for (int t = 0; t < 16; ++t) {
        const int e = (lane + 32 * t) * 4;
        *(float4*)(grow + e) = z4;
        *(float4*)(wrow + e) = z4;
      }
      // Scatter (stores retire in order per wave, so lands after the fill).
      const unsigned bits = vmask[r];
      const float wvf = wvals[r];
      if (lane < 16 && ((bits >> lane) & 1u)) {
        const unsigned j = selidx[r * 16 + lane];
        grow[j] = 1.0f;
        wrow[j] = wvf;
      }
    }
  }
}

extern "C" void kernel_launch(void* const* d_in, const int* in_sizes, int n_in,
                              void* d_out, int out_size, void* d_ws, size_t ws_size,
                              hipStream_t stream) {
  const float* s       = (const float*)d_in[0];
  const float* noise   = (const float*)d_in[1];
  const int*   adjp    = (const int*)d_in[2];
  const int*   activem = (const int*)d_in[3];
  const int*   actm    = (const int*)d_in[4];

  float* ctx  = (float*)d_out;
  float* gate = ctx + (size_t)BB * NN * DD;
  float* wout = gate + (size_t)BB * NN * NN;

  const size_t smem = (size_t)16 * NN * sizeof(float)     // scores
                    + (size_t)16 * 16 * sizeof(unsigned)  // selidx
                    + (size_t)16 * sizeof(unsigned)       // vmask
                    + (size_t)16 * sizeof(float);         // wvals

  dim3 grid(NN / 16, BB);
  rc_kernel<<<grid, 512, smem, stream>>>(s, noise, adjp, activem, actm,
                                         ctx, gate, wout);
  (void)in_sizes; (void)n_in; (void)out_size; (void)d_ws; (void)ws_size;
}
